// SSMLayer_17514876633683
// MI455X (gfx1250) — matrix operand order repaired and checked by reading, not compiled, not run on hardware
//
#include <hip/hip_runtime.h>
#include <hip/hip_bf16.h>
#include <math.h>

// ---------------------------------------------------------------------------
// SSM layer for MI455X (gfx1250, wave32, WMMA).
//
// Key algebraic collapse: the reference's state (B,S,H) is h-independent
// (input injection broadcast over h; A mixes only s), so the scan shrinks to
// st:(4,128) and its output to one scalar c_t[b].  Heavy work is two GEMMs
// done with v_wmma_f32_16x16x32_bf16 (f32 accumulate), epilogue fused.
// ---------------------------------------------------------------------------

typedef __attribute__((ext_vector_type(16))) __bf16 v16bf;
typedef __attribute__((ext_vector_type(8)))  float  v8f;

#define Hh   512
#define Ss   128
#define Tt   512
#define BTt  2048
#define LNE  1e-5f

// ---- workspace layout (bytes) ----
#define WS_XH   ((size_t)0)          // 2048*512 bf16  = 2,097,152
#define WS_DH   ((size_t)2097152)    // 512*512  bf16  =   524,288
#define WS_BH   ((size_t)2621440)    // 128*512  bf16  =   131,072
#define WS_U    ((size_t)2752512)    // 2048*128 f32   = 1,048,576
#define WS_CB   ((size_t)3801088)    // 128 f32
#define WS_C    ((size_t)3801600)    // 2048 f32 (c[t*4+b])

// ---------------- fragment loaders (ISA 05_wmma layouts) -------------------
// A fragment: 16x32 bf16, M = lane&15, VGPR i holds K pair
//   k0 = 2i + (i>=4 ? 8 : 0) + (lane>=16 ? 8 : 0)
__device__ __forceinline__ v16bf load_afrag(const __bf16* src, int ld,
                                            int mbase, int kbase, int lane) {
  v16bf a;
  const int hk = (lane >= 16) ? 8 : 0;
  const __bf16* row = src + (size_t)(mbase + (lane & 15)) * ld + kbase;
#pragma unroll
  for (int i = 0; i < 8; ++i) {
    int k0 = 2 * i + ((i >= 4) ? 8 : 0) + hk;
    a[2 * i]     = row[k0];
    a[2 * i + 1] = row[k0 + 1];
  }
  return a;
}

// B fragment: 32x16 bf16 (KxN), N = lane&15, VGPR i holds K pair
//   k0 = 2i + (i>=4 ? 8 : 0) + (lane>=16 ? 8 : 0).   Element [k][n] = srcT[n*ld + k]
// (srcT row-major N x K, i.e. D or B weight as stored -> computes x @ W^T).
__device__ __forceinline__ v16bf load_bfragT(const __bf16* srcT, int ld,
                                             int nbase, int kbase, int lane) {
  v16bf b;
  const int hk = (lane >= 16) ? 8 : 0;
  const __bf16* row = srcT + (size_t)(nbase + (lane & 15)) * ld + kbase;
#pragma unroll
  for (int i = 0; i < 8; ++i) {
    int k0 = 2 * i + ((i >= 4) ? 8 : 0) + hk;
    b[2 * i]     = row[k0];
    b[2 * i + 1] = row[k0 + 1];
  }
  return b;
}

// ------------------------- phase 0: conversions ----------------------------
__global__ void k_convert(const float* __restrict__ x, const float* __restrict__ D,
                          const float* __restrict__ Bm,
                          __bf16* __restrict__ xh, __bf16* __restrict__ Dh,
                          __bf16* __restrict__ Bh) {
  int i = blockIdx.x * blockDim.x + threadIdx.x;
  if (i < BTt * Hh) xh[i] = (__bf16)x[i];
  if (i < Hh * Hh)  Dh[i] = (__bf16)D[i];
  if (i < Ss * Hh)  Bh[i] = (__bf16)Bm[i];
}

__global__ void k_cbar(const float* __restrict__ C, float* __restrict__ Cbar) {
  int s = threadIdx.x;  // 128 threads, coalesced over rows of C (H x S)
  float acc = 0.f;
  for (int h = 0; h < Hh; ++h) acc += C[(size_t)h * Ss + s];
  Cbar[s] = acc * (1.0f / Hh);
}

// -------------------- phase 1: u = x @ B^T  (2048 x 128) -------------------
__global__ void k_u(const __bf16* __restrict__ xh, const __bf16* __restrict__ Bh,
                    float* __restrict__ u) {
  const int lane = threadIdx.x & 31;
  const int wid  = threadIdx.x >> 5;
  const int tile = blockIdx.x * 4 + wid;        // 1024 tiles: 128 M x 8 N
  const int mtile = tile >> 3, ntile = tile & 7;
  v8f acc = {};
  for (int ks = 0; ks < Hh; ks += 32) {
    v16bf a = load_afrag(xh, Hh, mtile * 16, ks, lane);
    v16bf b = load_bfragT(Bh, Hh, ntile * 16, ks, lane);
    acc = __builtin_amdgcn_wmma_f32_16x16x32_bf16(false, a, false, b,
                                                  (short)0, acc, false, false);
  }
  const int hi = (lane >= 16) ? 8 : 0;
  const int n  = ntile * 16 + (lane & 15);
#pragma unroll
  for (int r = 0; r < 8; ++r) {
    int m = mtile * 16 + r + hi;
    u[(size_t)m * Ss + n] = acc[r];
  }
}

// ------------- phase 2: serial scan, st:(4,128), emits c[t][b] -------------
// One block, 128 threads (thread p owns state row s=p for all 4 batches).
// A row A[p][*] cached in 128 VGPRs (fully unrolled); state in LDS as
// st[s][b] (float4 broadcast reads, conflict-free).
__global__ void k_scan(const float* __restrict__ A, const float* __restrict__ u,
                       const float* __restrict__ Cbar, float* __restrict__ cout) {
  __shared__ float st[Ss * 4];
  __shared__ float wred[4 * 4];
  const int p    = threadIdx.x;
  const int lane = p & 31, wid = p >> 5;

  float Areg[Ss];
#pragma unroll
  for (int s = 0; s < Ss; ++s) Areg[s] = A[(size_t)p * Ss + s];
  const float cb = Cbar[p];

  float sc0 = 0.f, sc1 = 0.f, sc2 = 0.f, sc3 = 0.f;  // st[p][b]
  st[p * 4 + 0] = 0.f; st[p * 4 + 1] = 0.f;
  st[p * 4 + 2] = 0.f; st[p * 4 + 3] = 0.f;
  __syncthreads();

  for (int t = 0; t < Tt; ++t) {
    // c_t[b] = Cbar . st_t[b]  (pre-update state) -- wave32 reductions
    float pc0 = cb * sc0, pc1 = cb * sc1, pc2 = cb * sc2, pc3 = cb * sc3;
#pragma unroll
    for (int off = 16; off >= 1; off >>= 1) {
      pc0 += __shfl_xor(pc0, off, 32);
      pc1 += __shfl_xor(pc1, off, 32);
      pc2 += __shfl_xor(pc2, off, 32);
      pc3 += __shfl_xor(pc3, off, 32);
    }
    if (lane == 0) {
      wred[wid * 4 + 0] = pc0; wred[wid * 4 + 1] = pc1;
      wred[wid * 4 + 2] = pc2; wred[wid * 4 + 3] = pc3;
    }

    // new_st[p][b] = sum_s A[p][s] * st[s][b] + u[b*T+t][p]
    float a0 = u[((size_t)(0 * Tt + t)) * Ss + p];
    float a1 = u[((size_t)(1 * Tt + t)) * Ss + p];
    float a2 = u[((size_t)(2 * Tt + t)) * Ss + p];
    float a3 = u[((size_t)(3 * Tt + t)) * Ss + p];
#pragma unroll
    for (int s = 0; s < Ss; ++s) {
      float4 sv = *(const float4*)&st[s * 4];
      float av = Areg[s];
      a0 = fmaf(av, sv.x, a0);
      a1 = fmaf(av, sv.y, a1);
      a2 = fmaf(av, sv.z, a2);
      a3 = fmaf(av, sv.w, a3);
    }
    __syncthreads();  // wred written; old-state reads done
    if (threadIdx.x < 4) {
      cout[t * 4 + threadIdx.x] = wred[0 + threadIdx.x] + wred[4 + threadIdx.x] +
                                  wred[8 + threadIdx.x] + wred[12 + threadIdx.x];
    }
    st[p * 4 + 0] = a0; st[p * 4 + 1] = a1;
    st[p * 4 + 2] = a2; st[p * 4 + 3] = a3;
    sc0 = a0; sc1 = a1; sc2 = a2; sc3 = a3;
    __syncthreads();  // new state visible
  }
}

// ------ phase 3: y = LN(gelu(c + x@D^T) + x) * gamma + beta, fused ---------
// 128 blocks, each: 16-row strip x 512 cols; 8 waves x 4 WMMA tiles each.
__global__ void k_final(const __bf16* __restrict__ xh, const __bf16* __restrict__ Dh,
                        const float* __restrict__ x, const float* __restrict__ cvec,
                        const float* __restrict__ gamma, const float* __restrict__ beta,
                        float* __restrict__ out) {
  __shared__ float yt[16 * 516];  // pitch 516 breaks row-pair bank aliasing
  const int lane  = threadIdx.x & 31;
  const int wid   = threadIdx.x >> 5;  // 8 waves
  const int mbase = blockIdx.x * 16;

  v8f zero = {0.f, 0.f, 0.f, 0.f, 0.f, 0.f, 0.f, 0.f};
  v8f acc[4];
#pragma unroll
  for (int j = 0; j < 4; ++j) acc[j] = zero;

  for (int ks = 0; ks < Hh; ks += 32) {
    v16bf a = load_afrag(xh, Hh, mbase, ks, lane);
#pragma unroll
    for (int j = 0; j < 4; ++j) {
      v16bf b = load_bfragT(Dh, Hh, (wid * 4 + j) * 16, ks, lane);
      acc[j] = __builtin_amdgcn_wmma_f32_16x16x32_bf16(false, a, false, b,
                                                       (short)0, acc[j], false, false);
    }
  }

  const int hi  = (lane >= 16) ? 8 : 0;
  const int nin = lane & 15;
  const float inv_sqrt2 = 0.70710678118654752f;
#pragma unroll
  for (int r = 0; r < 8; ++r) {
    const int m = mbase + r + hi;
    const int b = m >> 9, t = m & 511;       // rows are (b,t) with b major
    const float cv = cvec[t * 4 + b];
#pragma unroll
    for (int j = 0; j < 4; ++j) {
      int n = (wid * 4 + j) * 16 + nin;
      float v = acc[j][r] + cv;
      float g = 0.5f * v * (1.0f + erff(v * inv_sqrt2));  // exact GELU
      yt[(r + hi) * 516 + n] = g + x[(size_t)m * Hh + n]; // residual
    }
  }
  __syncthreads();

  // LayerNorm: wave wid handles rows 2*wid, 2*wid+1 (16 cols per lane)
#pragma unroll
  for (int rr2 = 0; rr2 < 2; ++rr2) {
    const int rr = wid * 2 + rr2;
    const int m  = mbase + rr;
    float s1 = 0.f, s2 = 0.f;
#pragma unroll
    for (int i = 0; i < 16; ++i) {
      float yv = yt[rr * 516 + lane + i * 32];
      s1 += yv; s2 += yv * yv;
    }
#pragma unroll
    for (int off = 16; off >= 1; off >>= 1) {
      s1 += __shfl_xor(s1, off, 32);
      s2 += __shfl_xor(s2, off, 32);
    }
    const float mean = s1 * (1.0f / Hh);
    const float var  = s2 * (1.0f / Hh) - mean * mean;
    const float rstd = rsqrtf(var + LNE);
#pragma unroll
    for (int i = 0; i < 16; ++i) {
      int col = lane + i * 32;
      float yv = yt[rr * 516 + col];
      out[(size_t)m * Hh + col] = (yv - mean) * rstd * gamma[col] + beta[col];
    }
  }
}

// ------------------------------- launcher ----------------------------------
extern "C" void kernel_launch(void* const* d_in, const int* in_sizes, int n_in,
                              void* d_out, int out_size, void* d_ws, size_t ws_size,
                              hipStream_t stream) {
  const float* x     = (const float*)d_in[0];
  const float* A     = (const float*)d_in[1];
  const float* Bm    = (const float*)d_in[2];
  const float* C     = (const float*)d_in[3];
  const float* D     = (const float*)d_in[4];
  const float* gamma = (const float*)d_in[5];
  const float* beta  = (const float*)d_in[6];

  char* ws = (char*)d_ws;
  __bf16* xh   = (__bf16*)(ws + WS_XH);
  __bf16* Dh   = (__bf16*)(ws + WS_DH);
  __bf16* Bh   = (__bf16*)(ws + WS_BH);
  float*  u    = (float*)(ws + WS_U);
  float*  Cbar = (float*)(ws + WS_CB);
  float*  cvec = (float*)(ws + WS_C);
  float*  out  = (float*)d_out;

  k_convert<<<4096, 256, 0, stream>>>(x, D, Bm, xh, Dh, Bh);
  k_cbar<<<1, 128, 0, stream>>>(C, Cbar);
  k_u<<<256, 128, 0, stream>>>(xh, Bh, u);
  k_scan<<<1, 128, 0, stream>>>(A, u, Cbar, cvec);
  k_final<<<128, 256, 0, stream>>>(xh, Dh, x, cvec, gamma, beta, out);
}
